// GAT_82197084111388
// MI455X (gfx1250) — matrix operand order repaired and checked by reading, not compiled, not run on hardware
//
#include <hip/hip_runtime.h>

// ---------------------------------------------------------------------------
// GAT forward for MI455X (gfx1250): bf16 WMMA GEMMs + L2-resident edge scatter
// ---------------------------------------------------------------------------

typedef __bf16 bf16_t;
typedef __attribute__((ext_vector_type(16))) __bf16 v16bf;
typedef __attribute__((ext_vector_type(8)))  __bf16 v8bf;
typedef __attribute__((ext_vector_type(8)))  float  v8f;

#define NEG_SLOPE 0.2f

__device__ __forceinline__ unsigned short f2bf(float f) {
  unsigned int u = __float_as_uint(f);
  unsigned int r = u + 0x7FFFu + ((u >> 16) & 1u);   // round-to-nearest-even
  return (unsigned short)(r >> 16);
}

__device__ __forceinline__ void atomicMaxF32(float* addr, float v) {
  // monotone-bits trick: works for mixed-sign floats, init must be very negative
  if (v >= 0.0f) atomicMax((int*)addr, __float_as_int(v));
  else           atomicMin((unsigned int*)addr, __float_as_uint(v));
}

// ------------------------------- elementwise -------------------------------

__global__ __launch_bounds__(256) void k_f32_to_bf16(
    const float* __restrict__ in, unsigned short* __restrict__ out, int n) {
  int i = blockIdx.x * blockDim.x + threadIdx.x;
  if (i < n) out[i] = f2bf(in[i]);
}

__global__ __launch_bounds__(256) void k_fill_f32(
    float* __restrict__ p, int n, float v) {
  int i = blockIdx.x * blockDim.x + threadIdx.x;
  if (i < n) p[i] = v;
}

// W [K,M] fp32  ->  Bt [Mpad,K] bf16 (zero-padded columns)
__global__ __launch_bounds__(256) void k_wT_bf16(
    const float* __restrict__ W, unsigned short* __restrict__ Bt,
    int K, int M, int Mpad) {
  int i = blockIdx.x * blockDim.x + threadIdx.x;
  if (i >= Mpad * K) return;
  int c = i / K;
  int r = i - c * K;
  float v = (c < M) ? W[(size_t)r * M + c] : 0.0f;
  Bt[i] = f2bf(v);
}

// ------------------------------ WMMA GEMM ----------------------------------
// C[N,M] (fp32) = A[N,K] (bf16, row-major) x Bt[Mpad,K]^T (bf16)
// 256 threads = 8 waves; wave tile 16x16; block tile 64x32.
// Requires N % 16 == 0 (true: N = 10000), K % 32 == 0, grid.y = Mpad/32.
__global__ __launch_bounds__(256) void k_gemm_bf16_wmma(
    const bf16_t* __restrict__ A, const bf16_t* __restrict__ Bt,
    float* __restrict__ C, int Nrows, int K, int M) {
  const int lane = threadIdx.x & 31;
  const int wave = threadIdx.x >> 5;
  const int wr = wave >> 1;                 // 0..3
  const int wc = wave & 1;                  // 0..1
  const int r0 = (blockIdx.x * 4 + wr) * 16;
  const int c0 = (blockIdx.y * 2 + wc) * 16;
  if (r0 >= Nrows) return;                  // wave-uniform: EXEC stays all-1s

  const int row   = r0 + (lane & 15);
  const int col   = c0 + (lane & 15);
  const int abase = (lane < 16) ? 0 : 8;    // ISA 16-bit A 16x32 layout
  const int bbase = (lane < 16) ? 0 : 8;    // same k order as A

  const bf16_t* __restrict__ Arow = A  + (size_t)row * K;
  const bf16_t* __restrict__ Brow = Bt + (size_t)col * K;

  v8f acc = {};
  for (int k0 = 0; k0 < K; k0 += 32) {
    union { v16bf v; v8bf h[2]; } a, b;
    a.h[0] = *(const v8bf*)(Arow + k0 + abase);        // K = base .. base+7
    a.h[1] = *(const v8bf*)(Arow + k0 + abase + 16);   // K = base+16 .. base+23
    b.h[0] = *(const v8bf*)(Brow + k0 + bbase);        // K = bbase .. bbase+7
    b.h[1] = *(const v8bf*)(Brow + k0 + bbase + 16);   // K = bbase+16 .. bbase+23
    acc = __builtin_amdgcn_wmma_f32_16x16x32_bf16(
        false, a.v, false, b.v, (short)0, acc, false, false);
  }

  const int rbase = r0 + ((lane < 16) ? 0 : 8);        // C/D layout
  if (col < M) {
#pragma unroll
    for (int r = 0; r < 8; ++r)
      C[(size_t)(rbase + r) * M + col] = acc[r];
  }
}

// --------------------------- attention pieces ------------------------------

// a_src[n,h] = sum_c xw[n,h,c]*att_src[h,c]; same for dst. One thread per (n,h).
__global__ __launch_bounds__(256) void k_att_coef(
    const float* __restrict__ xw, const float* __restrict__ att_s,
    const float* __restrict__ att_d, float* __restrict__ a_s,
    float* __restrict__ a_d, int Ncnt, int H, int Cc) {
  int i = blockIdx.x * blockDim.x + threadIdx.x;
  if (i >= Ncnt * H) return;
  int n = i / H, h = i - n * H;
  const float* xr = xw + (size_t)n * H * Cc + (size_t)h * Cc;
  const float* ws = att_s + h * Cc;
  const float* wd = att_d + h * Cc;
  float ss = 0.f, sd = 0.f;
  for (int c = 0; c < Cc; ++c) {
    float v = xr[c];
    ss += v * ws[c];
    sd += v * wd[c];
  }
  a_s[i] = ss;
  a_d[i] = sd;
}

__device__ __forceinline__ int edge_src(const int* ei, int E, int e) {
  return (e < E) ? ei[e] : (e - E);          // self loops appended
}
__device__ __forceinline__ int edge_dst(const int* ei, int E, int e) {
  return (e < E) ? ei[E + e] : (e - E);
}

// e = leaky_relu(a_src[s]+a_dst[d]); store; segment-max into m[d]
__global__ __launch_bounds__(256) void k_edge_logits(
    const int* __restrict__ ei, int E, int Etot,
    const float* __restrict__ a_s, const float* __restrict__ a_d, int H,
    float* __restrict__ evals, float* __restrict__ m) {
  int e = blockIdx.x * blockDim.x + threadIdx.x;
  if (e >= Etot) return;
  int s = edge_src(ei, E, e), d = edge_dst(ei, E, e);
  for (int h = 0; h < H; ++h) {
    float v = a_s[s * H + h] + a_d[d * H + h];
    v = (v > 0.f) ? v : NEG_SLOPE * v;
    evals[(size_t)e * H + h] = v;
    atomicMaxF32(&m[d * H + h], v);
  }
}

// ex = exp(e - m[d]); store; segment-sum into denom[d]
__global__ __launch_bounds__(256) void k_edge_exp(
    const int* __restrict__ ei, int E, int Etot, int H,
    const float* __restrict__ m, float* __restrict__ evals,
    float* __restrict__ denom) {
  int e = blockIdx.x * blockDim.x + threadIdx.x;
  if (e >= Etot) return;
  int d = edge_dst(ei, E, e);
  for (int h = 0; h < H; ++h) {
    float ex = __expf(evals[(size_t)e * H + h] - m[d * H + h]);
    evals[(size_t)e * H + h] = ex;
    atomicAdd(&denom[d * H + h], ex);
  }
}

// agg[d,f] += xw[s,f] * alpha[e, f/Cc]; one wave per edge, lanes stream F.
__global__ __launch_bounds__(256) void k_edge_aggregate(
    const int* __restrict__ ei, int E, int Etot, int H, int Cc, int F,
    const float* __restrict__ xw, const float* __restrict__ evals,
    const float* __restrict__ denom, float* __restrict__ agg) {
  int tid  = blockIdx.x * blockDim.x + threadIdx.x;
  int e    = tid >> 5;
  int lane = tid & 31;
  if (e >= Etot) return;
  int s = edge_src(ei, E, e), d = edge_dst(ei, E, e);
  const float* __restrict__ xr = xw + (size_t)s * F;
  float* __restrict__ ar = agg + (size_t)d * F;
  for (int f = lane; f < F; f += 32) {
    int h = f / Cc;
    float alpha = evals[(size_t)e * H + h] / (denom[d * H + h] + 1e-16f);
    atomicAdd(&ar[f], xr[f] * alpha);
  }
}

// x_next = elu(agg + bias + skip); also emit bf16 copy for the next GEMM
__global__ __launch_bounds__(256) void k_post_elu(
    const float* __restrict__ agg, const float* __restrict__ bias,
    const float* __restrict__ skip, float* __restrict__ xo,
    unsigned short* __restrict__ xob, int n, int F) {
  int i = blockIdx.x * blockDim.x + threadIdx.x;
  if (i >= n) return;
  float v = agg[i] + bias[i % F] + skip[i];
  v = (v > 0.f) ? v : (__expf(v) - 1.0f);
  xo[i] = v;
  xob[i] = f2bf(v);
}

// out = mean_h(agg) + b2 + skipout
__global__ __launch_bounds__(256) void k_final(
    const float* __restrict__ agg, const float* __restrict__ b2,
    const float* __restrict__ skipout, float* __restrict__ out,
    int Ncnt, int H, int Cls) {
  int i = blockIdx.x * blockDim.x + threadIdx.x;
  if (i >= Ncnt * Cls) return;
  int n = i / Cls, k = i - n * Cls;
  float s = 0.f;
  for (int h = 0; h < H; ++h) s += agg[(size_t)n * H * Cls + h * Cls + k];
  out[i] = s * (1.0f / (float)H) + b2[k] + skipout[i];
}

// ------------------------------- launcher ----------------------------------

static inline unsigned g1(long long n) { return (unsigned)((n + 255) / 256); }

extern "C" void kernel_launch(void* const* d_in, const int* in_sizes, int n_in,
                              void* d_out, int out_size, void* d_ws, size_t ws_size,
                              hipStream_t stream) {
  const float* x       = (const float*)d_in[0];
  const int*   ei      = (const int*)d_in[1];
  const float* W0      = (const float*)d_in[2];
  const float* a_src0  = (const float*)d_in[3];
  const float* a_dst0  = (const float*)d_in[4];
  const float* b0      = (const float*)d_in[5];
  const float* Wskipin = (const float*)d_in[6];
  const float* W1      = (const float*)d_in[7];
  const float* a_src1  = (const float*)d_in[8];
  const float* a_dst1  = (const float*)d_in[9];
  const float* b1      = (const float*)d_in[10];
  const float* W2      = (const float*)d_in[11];
  const float* a_src2  = (const float*)d_in[12];
  const float* a_dst2  = (const float*)d_in[13];
  const float* b2      = (const float*)d_in[14];
  const float* Wskipo  = (const float*)d_in[15];

  const int N    = in_sizes[0] / 256;     // 10000
  const int E    = in_sizes[1] / 2;       // 160000
  const int Etot = E + N;                 // +self loops
  const int FEAT = 256, HID = 512, H = 4, C = 128, H2 = 6, C2 = 40, F2 = H2 * C2;

  // ---- workspace carve-up ----
  size_t off = 0;
  char* base = (char*)d_ws;
  auto alloc = [&](size_t bytes) -> void* {
    void* p = base + off;
    off += (bytes + 255) & ~(size_t)255;
    return p;
  };
  unsigned short* xb   = (unsigned short*)alloc((size_t)N * FEAT * 2);
  unsigned short* W0t  = (unsigned short*)alloc((size_t)512 * 256 * 2);
  unsigned short* WsiT = (unsigned short*)alloc((size_t)512 * 256 * 2);
  unsigned short* W1t  = (unsigned short*)alloc((size_t)512 * 512 * 2);
  unsigned short* W2t  = (unsigned short*)alloc((size_t)256 * 512 * 2);   // Mpad=256
  unsigned short* WsoT = (unsigned short*)alloc((size_t)64  * 512 * 2);   // Mpad=64
  float* xw      = (float*)alloc((size_t)N * HID * 4);
  float* agg     = (float*)alloc((size_t)N * HID * 4);
  float* skip_in = (float*)alloc((size_t)N * HID * 4);
  float* skipout = (float*)alloc((size_t)N * 40 * 4);
  float* x1      = (float*)alloc((size_t)N * HID * 4);
  unsigned short* x1b = (unsigned short*)alloc((size_t)N * HID * 2);
  float* x2      = (float*)alloc((size_t)N * HID * 4);
  unsigned short* x2b = (unsigned short*)alloc((size_t)N * HID * 2);
  float* a_s   = (float*)alloc((size_t)N * 6 * 4);
  float* a_d   = (float*)alloc((size_t)N * 6 * 4);
  float* mbuf  = (float*)alloc((size_t)N * 6 * 4);
  float* dbuf  = (float*)alloc((size_t)N * 6 * 4);
  float* evals = (float*)alloc((size_t)Etot * 6 * 4);

  const dim3 B(256);

  // ---- precision conversion + weight transpose (tiny) ----
  k_f32_to_bf16<<<g1((long long)N * FEAT), B, 0, stream>>>(x, xb, N * FEAT);
  k_wT_bf16<<<g1(512 * 256), B, 0, stream>>>(W0,      W0t,  256, 512, 512);
  k_wT_bf16<<<g1(512 * 256), B, 0, stream>>>(Wskipin, WsiT, 256, 512, 512);
  k_wT_bf16<<<g1(512 * 512), B, 0, stream>>>(W1,      W1t,  512, 512, 512);
  k_wT_bf16<<<g1(256 * 512), B, 0, stream>>>(W2,      W2t,  512, 240, 256);
  k_wT_bf16<<<g1(64  * 512), B, 0, stream>>>(Wskipo,  WsoT, 512, 40,  64);

  const unsigned gr = (unsigned)((N + 63) / 64);

  // ================= layer 0 =================
  k_gemm_bf16_wmma<<<dim3(gr, 512 / 32), B, 0, stream>>>(
      (const bf16_t*)xb, (const bf16_t*)W0t, xw, N, 256, 512);
  k_gemm_bf16_wmma<<<dim3(gr, 512 / 32), B, 0, stream>>>(
      (const bf16_t*)xb, (const bf16_t*)WsiT, skip_in, N, 256, 512);
  k_att_coef<<<g1((long long)N * H), B, 0, stream>>>(xw, a_src0, a_dst0, a_s, a_d, N, H, C);
  k_fill_f32<<<g1((long long)N * H), B, 0, stream>>>(mbuf, N * H, -1e30f);
  k_fill_f32<<<g1((long long)N * H), B, 0, stream>>>(dbuf, N * H, 0.0f);
  k_fill_f32<<<g1((long long)N * HID), B, 0, stream>>>(agg, N * HID, 0.0f);
  k_edge_logits<<<g1(Etot), B, 0, stream>>>(ei, E, Etot, a_s, a_d, H, evals, mbuf);
  k_edge_exp<<<g1(Etot), B, 0, stream>>>(ei, E, Etot, H, mbuf, evals, dbuf);
  k_edge_aggregate<<<g1((long long)Etot * 32), B, 0, stream>>>(
      ei, E, Etot, H, C, HID, xw, evals, dbuf, agg);
  k_post_elu<<<g1((long long)N * HID), B, 0, stream>>>(agg, b0, skip_in, x1, x1b, N * HID, HID);

  // ================= layer 1 =================
  k_gemm_bf16_wmma<<<dim3(gr, 512 / 32), B, 0, stream>>>(
      (const bf16_t*)x1b, (const bf16_t*)W1t, xw, N, 512, 512);
  k_att_coef<<<g1((long long)N * H), B, 0, stream>>>(xw, a_src1, a_dst1, a_s, a_d, N, H, C);
  k_fill_f32<<<g1((long long)N * H), B, 0, stream>>>(mbuf, N * H, -1e30f);
  k_fill_f32<<<g1((long long)N * H), B, 0, stream>>>(dbuf, N * H, 0.0f);
  k_fill_f32<<<g1((long long)N * HID), B, 0, stream>>>(agg, N * HID, 0.0f);
  k_edge_logits<<<g1(Etot), B, 0, stream>>>(ei, E, Etot, a_s, a_d, H, evals, mbuf);
  k_edge_exp<<<g1(Etot), B, 0, stream>>>(ei, E, Etot, H, mbuf, evals, dbuf);
  k_edge_aggregate<<<g1((long long)Etot * 32), B, 0, stream>>>(
      ei, E, Etot, H, C, HID, xw, evals, dbuf, agg);
  k_post_elu<<<g1((long long)N * HID), B, 0, stream>>>(agg, b1, x1, x2, x2b, N * HID, HID);

  // ================= layer 2 =================
  k_gemm_bf16_wmma<<<dim3(gr, 256 / 32), B, 0, stream>>>(
      (const bf16_t*)x2b, (const bf16_t*)W2t, xw, N, 512, F2);
  k_gemm_bf16_wmma<<<dim3(gr, 64 / 32), B, 0, stream>>>(
      (const bf16_t*)x2b, (const bf16_t*)WsoT, skipout, N, 512, 40);
  k_att_coef<<<g1((long long)N * H2), B, 0, stream>>>(xw, a_src2, a_dst2, a_s, a_d, N, H2, C2);
  k_fill_f32<<<g1((long long)N * H2), B, 0, stream>>>(mbuf, N * H2, -1e30f);
  k_fill_f32<<<g1((long long)N * H2), B, 0, stream>>>(dbuf, N * H2, 0.0f);
  k_fill_f32<<<g1((long long)N * F2), B, 0, stream>>>(agg, N * F2, 0.0f);
  k_edge_logits<<<g1(Etot), B, 0, stream>>>(ei, E, Etot, a_s, a_d, H2, evals, mbuf);
  k_edge_exp<<<g1(Etot), B, 0, stream>>>(ei, E, Etot, H2, mbuf, evals, dbuf);
  k_edge_aggregate<<<g1((long long)Etot * 32), B, 0, stream>>>(
      ei, E, Etot, H2, C2, F2, xw, evals, dbuf, agg);
  k_final<<<g1((long long)N * 40), B, 0, stream>>>(agg, b2, skipout, (float*)d_out, N, H2, C2);
}
